// Generator_91122026152164
// MI455X (gfx1250) — compile-verified, matrix operand order checked by reading, not run on hardware
//
#include <hip/hip_runtime.h>
#include <hip/hip_bf16.h>
#include <math.h>

typedef _Float16 h16;
typedef __attribute__((ext_vector_type(16))) _Float16 v16h;
typedef __attribute__((ext_vector_type(8)))  float    v8f;
typedef int v4i_gnu __attribute__((vector_size(16)));   // pointee type of async-LDS builtin

// ---------------- problem constants ----------------
#define BATCH 2
#define CH    128
#define HH    80
#define WW    80
#define HS    40          // downsampled H
#define WSZ   40          // downsampled W
#define QN    1600        // HS*WSZ patch positions
#define K1    1152        // CH*3*3
#define K2    2048        // CH*4*4
#define EPSK  0.1152f     // EPS * K1 = 1e-4 * 1152
#define SCALE_F 10.0f
#define PI_F 3.14159265358979f

// ---------------- workspace layout (bytes) ----------------
#define OFF_LN 0
#define SZ_LN  ((size_t)2 * BATCH * QN * K1 * 2)            // 14,745,600
#define OFF_MM (OFF_LN + SZ_LN)
#define SZ_MM  ((size_t)BATCH * QN * K1 * 2)                // 7,372,800
#define OFF_AT OFF_LN                                       // overlay (20,480,000)
#define OFF_RP (OFF_MM + SZ_MM)                             // 22,118,400
#define SZ_RP  ((size_t)2 * BATCH * QN * K2 * 2)            // 26,214,400
#define OFF_S  (OFF_RP + SZ_RP)                             // 48,332,800
#define SZ_S   ((size_t)2 * BATCH * QN * QN * 4)            // 40,960,000
#define OFF_T  OFF_S                                        // overlay (26,214,400)
#define OFF_SF (OFF_S + SZ_S)                               // 89,292,800

// ---------------- CDNA5 async-to-LDS path (probe-gated) ----------------
#if defined(__HIP_DEVICE_COMPILE__) && \
    __has_builtin(__builtin_amdgcn_global_load_async_to_lds_b128) && \
    __has_builtin(__builtin_amdgcn_s_wait_asynccnt)
#define ASYNC_OK 1
#else
#define ASYNC_OK 0
#endif

#if ASYNC_OK
#define ASYNC_LD128(gp, lp)                                              \
    __builtin_amdgcn_global_load_async_to_lds_b128(                      \
        (__attribute__((address_space(1))) v4i_gnu*)(gp),                \
        (__attribute__((address_space(3))) v4i_gnu*)(lp), 0, 0)
#define WAIT_ASYNC0() __builtin_amdgcn_s_wait_asynccnt(0)
#endif

// =====================================================================
// Kernel 1: subsampled 3x3 patch matrices; left/right normalized, mid raw
// grid (QN, B, 3) block 128;  z: 0=left(norm) 1=right(norm) 2=mid
// =====================================================================
__global__ __launch_bounds__(128) void k_patch(const float* __restrict__ left,
                                               const float* __restrict__ right,
                                               const float* __restrict__ mid,
                                               h16* __restrict__ LN,
                                               h16* __restrict__ MM) {
    const int q = blockIdx.x, b = blockIdx.y, z = blockIdx.z;
    const float* src = (z == 0) ? left : (z == 1) ? right : mid;
    const int hq = q / WSZ, wq = q % WSZ;
    const int t = threadIdx.x;

    float vals[9];
    float ss = 0.f;
#pragma unroll
    for (int i = 0; i < 9; ++i) {
        int k = t + i * 128;                 // k in [0, 1152)
        int c  = k / 9;
        int ij = k % 9;
        int ii = ij / 3, jj = ij % 3;
        int hs = hq + ii - 1, ws = wq + jj - 1;   // 40-grid coords, pad=1
        float v = 0.f;
        if (hs >= 0 && hs < HS && ws >= 0 && ws < WSZ)
            v = src[(((size_t)b * CH + c) * HH + 2 * hs) * WW + 2 * ws];
        vals[i] = v;
        ss += v * v;
    }
    __shared__ float red[128];
    red[t] = ss;
    __syncthreads();
    for (int s = 64; s > 0; s >>= 1) {
        if (t < s) red[t] += red[t + s];
        __syncthreads();
    }
    const float nrm = (z < 2) ? rsqrtf(red[0] + EPSK) : 1.0f;
    h16* dst = (z < 2) ? (LN + (((size_t)z * BATCH + b) * QN + q) * K1)
                       : (MM + ((size_t)b * QN + q) * K1);
#pragma unroll
    for (int i = 0; i < 9; ++i)
        dst[t + i * 128] = (h16)(vals[i] * nrm);
}

// =====================================================================
// Kernel 2: 4x4 stride-2 raw patches (pad 1), f16.  grid (QN, B, 2) block 256
// =====================================================================
__global__ __launch_bounds__(256) void k_rawpatch(const float* __restrict__ rawl,
                                                  const float* __restrict__ rawr,
                                                  h16* __restrict__ RP) {
    const int q = blockIdx.x, b = blockIdx.y, s = blockIdx.z;
    const float* src = (s == 0) ? rawl : rawr;
    const int hq = q / WSZ, wq = q % WSZ;
    h16* dst = RP + (((size_t)s * BATCH + b) * QN + q) * K2;
#pragma unroll
    for (int i = 0; i < 8; ++i) {
        int ck = threadIdx.x + i * 256;      // [0, 2048)
        int c  = ck >> 4;
        int ab = ck & 15;
        int a  = ab >> 2, bb = ab & 3;
        int ys = 2 * hq - 1 + a, xs = 2 * wq - 1 + bb;
        float v = 0.f;
        if (ys >= 0 && ys < HH && xs >= 0 && xs < WW)
            v = src[(((size_t)b * CH + c) * HH + ys) * WW + xs];
        dst[ck] = (h16)v;
    }
}

// =====================================================================
// WMMA fragment loaders (per CDNA5 ISA 16-bit A-matrix 16x32 layout):
// lane L holds row M = L&15; half h maps to K = (h<8 ? h : h+8) + (L>=16 ? 8:0)
// =====================================================================
__device__ __forceinline__ v16h load_frag_rowK(const h16* lds, int row, int ldk, int lane) {
    v16h f;
    const h16* pr = lds + (size_t)(row + (lane & 15)) * ldk + ((lane >> 4) << 3);
#pragma unroll
    for (int h = 0; h < 8; ++h)  f[h] = pr[h];          // K = kb + 0..7
#pragma unroll
    for (int h = 8; h < 16; ++h) f[h] = pr[h + 8];      // K = kb + 16..23
    return f;
}

__device__ __forceinline__ v16h load_frag_KN(const h16* lds, int col, int ldn, int lane) {
    v16h f;
    const int n  = col + (lane & 15);
    const int kb = (lane >> 4) << 3;
#pragma unroll
    for (int h = 0; h < 8; ++h)  f[h] = lds[(size_t)(kb + h) * ldn + n];
#pragma unroll
    for (int h = 8; h < 16; ++h) f[h] = lds[(size_t)(kb + 16 + (h - 8)) * ldn + n];
    return f;
}

#define LDKP 40   // 32 + 8 halves pad -> conflict-free 16B lane reads
#define LDN2 136

#if ASYNC_OK
// async DMA stage of a [128 rows x 32 k] K-contiguous tile (rows guarded)
__device__ __forceinline__ void stage_rowK(const h16* src, h16* dst, int row0,
                                           int ldg, int kt, int tid) {
#pragma unroll
    for (int it = 0; it < 2; ++it) {
        int chunk = tid + it * 256;          // 512 x 16B chunks
        int r  = chunk >> 2;
        int c8 = (chunk & 3) * 8;
        if (row0 + r < QN)
            ASYNC_LD128(src + (size_t)(row0 + r) * ldg + kt + c8,
                        dst + (size_t)r * LDKP + c8);
    }
}
// async DMA stage of a [32 k-rows x 128 cols] tile (always in range)
__device__ __forceinline__ void stage_KN(const h16* src, h16* dst, int tn,
                                         int kt, int tid) {
#pragma unroll
    for (int it = 0; it < 2; ++it) {
        int chunk = tid + it * 256;
        int r  = chunk >> 4;
        int c8 = (chunk & 15) * 8;
        ASYNC_LD128(src + (size_t)(kt + r) * K2 + tn + c8,
                    dst + (size_t)r * LDN2 + c8);
    }
}
#endif

// =====================================================================
// Kernel 3: GEMM-NT  S[q,p] = sum_k LN[q,k]*MM[p,k]
// grid (13,13,4) block 256 (8 waves, 128x128 block tile, 32x64 wave tile)
// =====================================================================
__global__ __launch_bounds__(256) void k_gemm_nt(const h16* __restrict__ Aall,
                                                 const h16* __restrict__ Ball,
                                                 float* __restrict__ Call) {
    const int sb = blockIdx.z;            // side*B + b
    const int b  = sb % BATCH;
    const h16* A  = Aall + (size_t)sb * QN * K1;
    const h16* Bm = Ball + (size_t)b  * QN * K1;
    float*     Cc = Call + (size_t)sb * QN * QN;

    const int tid  = threadIdx.x;
    const int lane = tid & 31, wid = tid >> 5;
    const int wm0 = (wid & 3) * 32, wn0 = (wid >> 2) * 64;
    const int tm = blockIdx.x * 128, tn = blockIdx.y * 128;

    v8f acc[2][4] = {};

#if ASYNC_OK
    // -------- CDNA5 async-to-LDS, double buffered, 1 barrier / K-step ----
    __shared__ h16 As[2][128][LDKP];
    __shared__ h16 Bs[2][128][LDKP];

    // pre-zero OOB rows (last M/N tile); async loads never touch them
#pragma unroll
    for (int it = 0; it < 2; ++it) {
        int chunk = tid + it * 256;
        int r  = chunk >> 2;
        int c8 = (chunk & 3) * 8;
        uint4 z = {0, 0, 0, 0};
        if (tm + r >= QN) { *(uint4*)&As[0][r][c8] = z; *(uint4*)&As[1][r][c8] = z; }
        if (tn + r >= QN) { *(uint4*)&Bs[0][r][c8] = z; *(uint4*)&Bs[1][r][c8] = z; }
    }
    stage_rowK(A,  &As[0][0][0], tm, K1, 0, tid);
    stage_rowK(Bm, &Bs[0][0][0], tn, K1, 0, tid);
    WAIT_ASYNC0();
    __syncthreads();

    const int NK = K1 / 32;                       // 36 exact
    for (int ktile = 0; ktile < NK; ++ktile) {
        const int buf = ktile & 1;
        if (ktile + 1 < NK) {                     // DMA next tile during compute
            stage_rowK(A,  &As[buf ^ 1][0][0], tm, K1, (ktile + 1) * 32, tid);
            stage_rowK(Bm, &Bs[buf ^ 1][0][0], tn, K1, (ktile + 1) * 32, tid);
        }
        v16h af[2], bf[4];
#pragma unroll
        for (int mi = 0; mi < 2; ++mi) af[mi] = load_frag_rowK(&As[buf][0][0], wm0 + mi * 16, LDKP, lane);
#pragma unroll
        for (int ni = 0; ni < 4; ++ni) bf[ni] = load_frag_rowK(&Bs[buf][0][0], wn0 + ni * 16, LDKP, lane);
#pragma unroll
        for (int mi = 0; mi < 2; ++mi)
#pragma unroll
            for (int ni = 0; ni < 4; ++ni)
                acc[mi][ni] = __builtin_amdgcn_wmma_f32_16x16x32_f16(
                    false, af[mi], false, bf[ni], (short)0, acc[mi][ni], false, false);
        WAIT_ASYNC0();
        __syncthreads();
    }
#else
    // -------- fallback: register-staged, single buffer -------------------
    __shared__ h16 As[128][LDKP];
    __shared__ h16 Bs[128][LDKP];
    for (int kt = 0; kt < K1; kt += 32) {
#pragma unroll
        for (int it = 0; it < 2; ++it) {
            int chunk = tid + it * 256;
            int r  = chunk >> 2;
            int c8 = (chunk & 3) * 8;
            uint4 va = {0, 0, 0, 0}, vb = {0, 0, 0, 0};
            if (tm + r < QN) va = *reinterpret_cast<const uint4*>(A  + (size_t)(tm + r) * K1 + kt + c8);
            if (tn + r < QN) vb = *reinterpret_cast<const uint4*>(Bm + (size_t)(tn + r) * K1 + kt + c8);
            *reinterpret_cast<uint4*>(&As[r][c8]) = va;
            *reinterpret_cast<uint4*>(&Bs[r][c8]) = vb;
        }
        __syncthreads();
        v16h af[2], bf[4];
#pragma unroll
        for (int mi = 0; mi < 2; ++mi) af[mi] = load_frag_rowK(&As[0][0], wm0 + mi * 16, LDKP, lane);
#pragma unroll
        for (int ni = 0; ni < 4; ++ni) bf[ni] = load_frag_rowK(&Bs[0][0], wn0 + ni * 16, LDKP, lane);
#pragma unroll
        for (int mi = 0; mi < 2; ++mi)
#pragma unroll
            for (int ni = 0; ni < 4; ++ni)
                acc[mi][ni] = __builtin_amdgcn_wmma_f32_16x16x32_f16(
                    false, af[mi], false, bf[ni], (short)0, acc[mi][ni], false, false);
        __syncthreads();
    }
#endif

    const int rofs = (lane >> 4) << 3;  // upper half-wave holds rows +8
    const int col0 = lane & 15;
#pragma unroll
    for (int mi = 0; mi < 2; ++mi)
#pragma unroll
        for (int ni = 0; ni < 4; ++ni)
#pragma unroll
            for (int r = 0; r < 8; ++r) {
                int row = tm + wm0 + mi * 16 + r + rofs;
                int col = tn + wn0 + ni * 16 + col0;
                if (row < QN && col < QN)
                    Cc[(size_t)row * QN + col] = acc[mi][ni][r];
            }
}

// =====================================================================
// Kernel 4: diagonal-shift fuse (two flat-index 3x3 identity convs),
// written TRANSPOSED: SF[sb][p][q] = fused(q,p)
// =====================================================================
__global__ void k_fuse(const float* __restrict__ Sb, float* __restrict__ SF) {
    size_t gid = (size_t)blockIdx.x * blockDim.x + threadIdx.x;
    if (gid >= (size_t)2 * BATCH * QN * QN) return;
    const int q  = (int)(gid % QN);
    const int p  = (int)((gid / QN) % QN);
    const int sb = (int)(gid / ((size_t)QN * QN));
    const float* Sp = Sb + (size_t)sb * QN * QN;

    const int hl = q / WSZ, wl = q % WSZ;
    const int hm = p / WSZ, wm = p % WSZ;
    const int q2 = wl * HS + hl, p2 = wm * HS + hm;   // transposed flat coords

    float acc = 0.f;
#pragma unroll
    for (int d2 = -1; d2 <= 1; ++d2) {
        int q2d = q2 + d2, p2d = p2 + d2;
        if (q2d < 0 || q2d >= QN || p2d < 0 || p2d >= QN) continue;
        int wl2 = q2d / HS, hl2 = q2d % HS;
        int wm2 = p2d / HS, hm2 = p2d % HS;
        int qb = hl2 * WSZ + wl2, pb = hm2 * WSZ + wm2;
#pragma unroll
        for (int d1 = -1; d1 <= 1; ++d1) {
            int qq = qb + d1, pp = pb + d1;
            if (qq < 0 || qq >= QN || pp < 0 || pp >= QN) continue;
            acc += Sp[(size_t)qq * QN + pp];
        }
    }
    SF[((size_t)sb * QN + p) * QN + q] = acc;
}

// =====================================================================
// Kernel 5: softmax over q (contiguous rows of SF), write f16 attn^T
// grid (QN, 4) block 256
// =====================================================================
__global__ __launch_bounds__(256) void k_softmax(const float* __restrict__ SF,
                                                 h16* __restrict__ AT) {
    const int p = blockIdx.x, sb = blockIdx.y;
    const float* row = SF + ((size_t)sb * QN + p) * QN;
    h16* orow = AT + ((size_t)sb * QN + p) * QN;
    const int t = threadIdx.x;
    __shared__ float red[256];

    float mx = -1e30f;
    for (int k = t; k < QN; k += 256) mx = fmaxf(mx, row[k]);
    red[t] = mx; __syncthreads();
    for (int s = 128; s > 0; s >>= 1) { if (t < s) red[t] = fmaxf(red[t], red[t + s]); __syncthreads(); }
    mx = red[0]; __syncthreads();

    float sum = 0.f;
    for (int k = t; k < QN; k += 256) sum += __expf(SCALE_F * (row[k] - mx));
    red[t] = sum; __syncthreads();
    for (int s = 128; s > 0; s >>= 1) { if (t < s) red[t] += red[t + s]; __syncthreads(); }
    const float inv = 1.0f / red[0];

    for (int k = t; k < QN; k += 256)
        orow[k] = (h16)(__expf(SCALE_F * (row[k] - mx)) * inv);
}

// =====================================================================
// Kernel 6: GEMM-NN  T[p,ck] = sum_q AT[p,q] * RP[q,ck]   (f16 out)
// grid (13,16,4) block 256
// =====================================================================
__global__ __launch_bounds__(256) void k_gemm_nn(const h16* __restrict__ Aall,
                                                 const h16* __restrict__ Ball,
                                                 h16* __restrict__ Call) {
    const int sb = blockIdx.z;
    const h16* A  = Aall + (size_t)sb * QN * QN;   // attn^T [p][q]
    const h16* Bb = Ball + (size_t)sb * QN * K2;   // rawp   [q][ck]
    h16*       Cc = Call + (size_t)sb * QN * K2;

    const int tid  = threadIdx.x;
    const int lane = tid & 31, wid = tid >> 5;
    const int wm0 = (wid & 3) * 32, wn0 = (wid >> 2) * 64;
    const int tm = blockIdx.x * 128, tn = blockIdx.y * 128;

    v8f acc[2][4] = {};

#if ASYNC_OK
    __shared__ h16 As[2][128][LDKP];
    __shared__ h16 Bs[2][32][LDN2];

#pragma unroll
    for (int it = 0; it < 2; ++it) {
        int chunk = tid + it * 256;
        int r  = chunk >> 2;
        int c8 = (chunk & 3) * 8;
        uint4 z = {0, 0, 0, 0};
        if (tm + r >= QN) { *(uint4*)&As[0][r][c8] = z; *(uint4*)&As[1][r][c8] = z; }
    }
    stage_rowK(A, &As[0][0][0], tm, QN, 0, tid);
    stage_KN(Bb, &Bs[0][0][0], tn, 0, tid);
    WAIT_ASYNC0();
    __syncthreads();

    const int NK = QN / 32;                       // 50 exact
    for (int ktile = 0; ktile < NK; ++ktile) {
        const int buf = ktile & 1;
        if (ktile + 1 < NK) {
            stage_rowK(A, &As[buf ^ 1][0][0], tm, QN, (ktile + 1) * 32, tid);
            stage_KN(Bb, &Bs[buf ^ 1][0][0], tn, (ktile + 1) * 32, tid);
        }
        v16h af[2], bf[4];
#pragma unroll
        for (int mi = 0; mi < 2; ++mi) af[mi] = load_frag_rowK(&As[buf][0][0], wm0 + mi * 16, LDKP, lane);
#pragma unroll
        for (int ni = 0; ni < 4; ++ni) bf[ni] = load_frag_KN(&Bs[buf][0][0], wn0 + ni * 16, LDN2, lane);
#pragma unroll
        for (int mi = 0; mi < 2; ++mi)
#pragma unroll
            for (int ni = 0; ni < 4; ++ni)
                acc[mi][ni] = __builtin_amdgcn_wmma_f32_16x16x32_f16(
                    false, af[mi], false, bf[ni], (short)0, acc[mi][ni], false, false);
        WAIT_ASYNC0();
        __syncthreads();
    }
#else
    __shared__ h16 As[128][LDKP];
    __shared__ h16 Bs[32][LDN2];
    for (int kt = 0; kt < QN; kt += 32) {
#pragma unroll
        for (int it = 0; it < 2; ++it) {
            int chunk = tid + it * 256;
            int ra  = chunk >> 2;
            int ca8 = (chunk & 3) * 8;
            uint4 va = {0, 0, 0, 0};
            if (tm + ra < QN) va = *reinterpret_cast<const uint4*>(A + (size_t)(tm + ra) * QN + kt + ca8);
            *reinterpret_cast<uint4*>(&As[ra][ca8]) = va;
            int rb  = chunk >> 4;
            int cb8 = (chunk & 15) * 8;
            uint4 vb = *reinterpret_cast<const uint4*>(Bb + (size_t)(kt + rb) * K2 + tn + cb8);
            *reinterpret_cast<uint4*>(&Bs[rb][cb8]) = vb;
        }
        __syncthreads();
        v16h af[2], bf[4];
#pragma unroll
        for (int mi = 0; mi < 2; ++mi) af[mi] = load_frag_rowK(&As[0][0], wm0 + mi * 16, LDKP, lane);
#pragma unroll
        for (int ni = 0; ni < 4; ++ni) bf[ni] = load_frag_KN(&Bs[0][0], wn0 + ni * 16, LDN2, lane);
#pragma unroll
        for (int mi = 0; mi < 2; ++mi)
#pragma unroll
            for (int ni = 0; ni < 4; ++ni)
                acc[mi][ni] = __builtin_amdgcn_wmma_f32_16x16x32_f16(
                    false, af[mi], false, bf[ni], (short)0, acc[mi][ni], false, false);
        __syncthreads();
    }
#endif

    const int rofs = (lane >> 4) << 3;
    const int col0 = lane & 15;
#pragma unroll
    for (int mi = 0; mi < 2; ++mi)
#pragma unroll
        for (int ni = 0; ni < 4; ++ni)
#pragma unroll
            for (int r = 0; r < 8; ++r) {
                int row = tm + wm0 + mi * 16 + r + rofs;
                int col = tn + wn0 + ni * 16 + col0;   // col < 2048 always
                if (row < QN)
                    Cc[(size_t)row * K2 + col] = (h16)acc[mi][ni][r];
            }
}

// =====================================================================
// Kernel 7: overlap-add gather (transposed-conv epilogue) + window blend
// =====================================================================
__global__ __launch_bounds__(256) void k_gather(const h16* __restrict__ Tb,
                                                float* __restrict__ out) {
    int idx = blockIdx.x * blockDim.x + threadIdx.x;
    if (idx >= BATCH * CH * HH * WW) return;
    const int x = idx % WW;
    const int y = (idx / WW) % HH;
    const int c = (idx / (WW * HH)) % CH;
    const int b = idx / (WW * HH * CH);

    const int hmc = (y + 1) >> 1;
    const int wmc = (x + 1) >> 1;

    float ys[2];
#pragma unroll
    for (int s = 0; s < 2; ++s) {
        const h16* Tp = Tb + ((size_t)(s * BATCH + b)) * QN * K2;
        float acc = 0.f;
#pragma unroll
        for (int dh = -1; dh <= 0; ++dh) {
            int hm = hmc + dh;
            int a  = y + 1 - 2 * hm;
            if (hm < 0 || hm >= HS || a < 0 || a > 3) continue;
#pragma unroll
            for (int dw = -1; dw <= 0; ++dw) {
                int wm = wmc + dw;
                int bb = x + 1 - 2 * wm;
                if (wm < 0 || wm >= WSZ || bb < 0 || bb > 3) continue;
                acc += (float)Tp[(size_t)(hm * WSZ + wm) * K2 + (c * 16 + a * 4 + bb)];
            }
        }
        ys[s] = acc * 0.25f;
    }
    const float wx  = 0.5f * (1.0f + __cosf(PI_F * (float)x / (float)(WW - 1)));
    const float wxf = 0.5f * (1.0f + __cosf(PI_F * (float)(WW - 1 - x) / (float)(WW - 1)));
    out[idx] = wx * ys[0] + wxf * ys[1];
}

// =====================================================================
extern "C" void kernel_launch(void* const* d_in, const int* in_sizes, int n_in,
                              void* d_out, int out_size, void* d_ws, size_t ws_size,
                              hipStream_t stream) {
    const float* left  = (const float*)d_in[0];
    const float* right = (const float*)d_in[1];
    const float* mid   = (const float*)d_in[2];
    const float* rawl  = (const float*)d_in[3];
    const float* rawr  = (const float*)d_in[4];
    float* out = (float*)d_out;

    char* ws = (char*)d_ws;
    h16*   LN = (h16*)  (ws + OFF_LN);
    h16*   MM = (h16*)  (ws + OFF_MM);
    h16*   RP = (h16*)  (ws + OFF_RP);
    float* S  = (float*)(ws + OFF_S);
    float* SF = (float*)(ws + OFF_SF);
    h16*   AT = (h16*)  (ws + OFF_AT);   // overlays LN+MM (dead after GEMM1)
    h16*   Tt = (h16*)  (ws + OFF_T);    // overlays S (dead after fuse)

    k_patch   <<<dim3(QN, BATCH, 3), 128, 0, stream>>>(left, right, mid, LN, MM);
    k_rawpatch<<<dim3(QN, BATCH, 2), 256, 0, stream>>>(rawl, rawr, RP);
    k_gemm_nt <<<dim3(13, 13, 2 * BATCH), 256, 0, stream>>>(LN, MM, S);

    const long fuse_total = (long)2 * BATCH * QN * QN;
    k_fuse    <<<(int)((fuse_total + 255) / 256), 256, 0, stream>>>(S, SF);
    k_softmax <<<dim3(QN, 2 * BATCH), 256, 0, stream>>>(SF, AT);
    k_gemm_nn <<<dim3(13, 16, 2 * BATCH), 256, 0, stream>>>(AT, RP, Tt);

    const int gtotal = BATCH * CH * HH * WW;
    k_gather  <<<(gtotal + 255) / 256, 256, 0, stream>>>(Tt, out);
}
